// GCoTNet_8933531976324
// MI455X (gfx1250) — compile-verified, matrix operand order checked by reading, not run on hardware
//
#include <hip/hip_runtime.h>
#include <math.h>

// ---------------------------------------------------------------------------
// GCoTNet forward for MI455X (gfx1250), wave32 + WMMA.
//
// Shapes (static): B=64 graphs, NP=256 nodes, EP=8192 edges, IN=64, H=128.
// N = 16384 nodes total, E = 524288 edges total.
//
// d_in layout (setup_inputs dict order; params flattened jax.tree order,
// i.e. sorted dict keys: blocks < causal < final; cot < gcn; a<c<ft<m;
// W<b; W1<W2<b1<b2<bt<g):
//   0 x[1048576] 1 row[E] 2 col[E] 3 largest[1]
//   4..17  block0: a.W a.b c.W c.b ft.W ft.b m.W m.b  W1 W2 b1 b2 bt g
//   18..31 block1: same
//   32 causal.W 33 causal.b
//   34..39 final: W1 W2 b1 b2 bt g
// d_out = [hg (64*128) | s0 (E) | s1 (E)]
// ---------------------------------------------------------------------------

#define NNODES 16384
#define HDIM   128
#define NEDGE  524288
#define NGRAPH 64
#define EPG    8192
#define BN_EPS 1e-5f

typedef __attribute__((ext_vector_type(16))) _Float16 v16h;
typedef __attribute__((ext_vector_type(8)))  float    v8f;

// ---------------- WMMA helpers (16x16x32 f16 -> f32) -----------------------

__device__ __forceinline__ v8f v8zero() {
  v8f c;
#pragma unroll
  for (int i = 0; i < 8; ++i) c[i] = 0.0f;
  return c;
}

__device__ __forceinline__ v8f wmma16(v16h a, v16h b, v8f c) {
  return __builtin_amdgcn_wmma_f32_16x16x32_f16(false, a, false, b, (short)0, c,
                                                false, false);
}

// K index held by half j of the A/B fragment (ISA 7.12.2, 16-bit 16x32 layout):
// VGPR i (halves 2i,2i+1): i<4 -> K = 2i + 8*sel, i>=4 -> K = 16 + 2(i-4) + 8*sel.
__device__ __forceinline__ int frag_kbase(int i, int sel) {
  return ((i < 4) ? (2 * i) : (16 + 2 * (i - 4))) + 8 * sel;
}

// Build A fragment (row m = lane&15 supplied via rowp) from a global f32 row,
// with per-row scale.  K pairs are even-aligned -> float2 loads.
__device__ __forceinline__ v16h load_a_global(const float* rowp, float scale) {
  int sel = (threadIdx.x >> 4) & 1;
  v16h a;
#pragma unroll
  for (int i = 0; i < 8; ++i) {
    int kb = frag_kbase(i, sel);
    float2 v = *(const float2*)(rowp + kb);
    a[2 * i]     = (_Float16)(v.x * scale);
    a[2 * i + 1] = (_Float16)(v.y * scale);
  }
  return a;
}

// Build A fragment from an LDS f32 row (64 wide), K offset kbase.
__device__ __forceinline__ v16h load_a_lds(const float* row64, int kbase) {
  int sel = (threadIdx.x >> 4) & 1;
  v16h a;
#pragma unroll
  for (int i = 0; i < 8; ++i) {
    int k = frag_kbase(i, sel) + kbase;
    a[2 * i]     = (_Float16)row64[k];
    a[2 * i + 1] = (_Float16)row64[k + 1];
  }
  return a;
}

// Fetch a pre-packed B fragment: 512 halves per (kt,nt) tile, 16 per lane.
__device__ __forceinline__ v16h load_b_pack(const _Float16* pack, int kt, int nt,
                                            int NT) {
  int lane = threadIdx.x & 31;
  return *(const v16h*)(pack + (((size_t)(kt * NT + nt)) << 9) + (lane << 4));
}

// ---------------- weight packing: f32 [K x N] row-major -> B fragments -----

__global__ void pack_b_kernel(const float* __restrict__ W, int ldn, int rowOff,
                              int K, int N, _Float16* __restrict__ out) {
  int total = (K >> 5) * (N >> 4) * 512;
  for (int t = blockIdx.x * blockDim.x + threadIdx.x; t < total;
       t += gridDim.x * blockDim.x) {
    int j = t & 15;
    int lane = (t >> 4) & 31;
    int fid = t >> 9;
    int NT = N >> 4;
    int kt = fid / NT, nt = fid - kt * NT;
    int sel = lane >> 4;
    int i = j >> 1, r = j & 1;
    int k = frag_kbase(i, sel) + r + 32 * kt;
    int n = (lane & 15) + 16 * nt;
    out[t] = (_Float16)W[(size_t)(rowOff + k) * ldn + n];
  }
}

// ---------------- generic utility kernels ----------------------------------

__global__ void fill_kernel(float* __restrict__ p, float v, long long n) {
  long long i = (long long)blockIdx.x * blockDim.x + threadIdx.x;
  if (i < n) p[i] = v;
}

__global__ void deg_kernel(const int* __restrict__ row, const int* __restrict__ col,
                           const float* __restrict__ active,
                           float* __restrict__ degO, float* __restrict__ degI,
                           int E) {
  int e = blockIdx.x * blockDim.x + threadIdx.x;
  if (e >= E) return;
  float a = active[e];
  if (a != 0.0f) {
    atomicAdd(&degO[row[e]], a);
    atomicAdd(&degI[col[e]], a);
  }
}

__global__ void invdeg_kernel(const float* __restrict__ degO,
                              const float* __restrict__ degI,
                              float* __restrict__ invO, float* __restrict__ invI,
                              float* __restrict__ degS, int N) {
  int i = blockIdx.x * blockDim.x + threadIdx.x;
  if (i >= N) return;
  float o = degO[i], in = degI[i];
  invO[i] = rsqrtf(fmaxf(o, 1.0f));
  invI[i] = rsqrtf(fmaxf(in, 1.0f));
  degS[i] = o + in;
}

// acc[col] += (h[row] * invO[row]) * ew, per channel.  One thread = (edge, 4ch).
__global__ void edge_agg_kernel(const float* __restrict__ h,
                                const int* __restrict__ row,
                                const int* __restrict__ col,
                                const float* __restrict__ ew,
                                const float* __restrict__ invO,
                                float* __restrict__ acc, int E) {
  long long tid = (long long)blockIdx.x * blockDim.x + threadIdx.x;
  if (tid >= (long long)E * 32) return;
  int e = (int)(tid >> 5);
  int cg = (int)(tid & 31);
  float w = ew[e];
  if (w == 0.0f) return;
  int r = row[e], c = col[e];
  float s = invO[r] * w;
  float4 hv = *(const float4*)(h + (size_t)r * HDIM + cg * 4);
  float* dst = acc + (size_t)c * HDIM + cg * 4;
  atomicAdd(dst + 0, s * hv.x);
  atomicAdd(dst + 1, s * hv.y);
  atomicAdd(dst + 2, s * hv.z);
  atomicAdd(dst + 3, s * hv.w);
}

// ---------------- node-level GEMM: out[16384 x 128] = (in*rowscale) @ W + b ---
// flags: bit0 = relu, bit1 = add resid after relu.
__global__ void __launch_bounds__(256)
gemm_node_kernel(const float* __restrict__ in, const float* __restrict__ rowscale,
                 const _Float16* __restrict__ wpack, const float* __restrict__ bias,
                 const float* __restrict__ resid, float* __restrict__ out, int K,
                 int flags) {
  const int NT = 8;  // N = 128
  int wv = threadIdx.x >> 5;   // wave -> N tile
  int lane = threadIdx.x & 31;
  int m = lane & 15, sel = lane >> 4;
  int mt = blockIdx.x;
  int gRowA = mt * 16 + m;
  float sc = rowscale ? rowscale[gRowA] : 1.0f;
  int KT = K >> 5;
  v8f c = v8zero();
  for (int kt = 0; kt < KT; ++kt) {
    v16h a = load_a_global(in + (size_t)gRowA * K + kt * 32, sc);
    v16h b = load_b_pack(wpack, kt, wv, NT);
    c = wmma16(a, b, c);
  }
  int n = wv * 16 + (lane & 15);
  float bv = bias[n];
#pragma unroll
  for (int r = 0; r < 8; ++r) {
    int gm = mt * 16 + r + 8 * sel;
    float v = c[r] + bv;
    if (flags & 1) v = fmaxf(v, 0.0f);
    if (flags & 2) v += resid[(size_t)gm * HDIM + n];
    out[(size_t)gm * HDIM + n] = v;
  }
}

// ---------------- BatchNorm (training-mode stats over all 16384 rows) ------

__global__ void __launch_bounds__(256)
bn_stats_kernel(const float* __restrict__ h, float* __restrict__ mu,
                float* __restrict__ var) {
  __shared__ float s1[256], s2[256];
  int c = blockIdx.x;
  float a = 0.0f, b = 0.0f;
  for (int r = threadIdx.x; r < NNODES; r += 256) {
    float v = h[(size_t)r * HDIM + c];
    a += v;
    b += v * v;
  }
  s1[threadIdx.x] = a;
  s2[threadIdx.x] = b;
  __syncthreads();
  for (int st = 128; st > 0; st >>= 1) {
    if (threadIdx.x < st) {
      s1[threadIdx.x] += s1[threadIdx.x + st];
      s2[threadIdx.x] += s2[threadIdx.x + st];
    }
    __syncthreads();
  }
  if (threadIdx.x == 0) {
    float mval = s1[0] / (float)NNODES;
    mu[c] = mval;
    var[c] = s2[0] / (float)NNODES - mval * mval;
  }
}

__global__ void bn_apply_kernel(const float* __restrict__ in,
                                const float* __restrict__ mu,
                                const float* __restrict__ var,
                                const float* __restrict__ g,
                                const float* __restrict__ bt,
                                float* __restrict__ out, long long n) {
  long long i = (long long)blockIdx.x * blockDim.x + threadIdx.x;
  if (i >= n) return;
  int c = (int)(i & (HDIM - 1));
  out[i] = (in[i] - mu[c]) * rsqrtf(var[c] + BN_EPS) * g[c] + bt[c];
}

// ---------------- CoT edge-score kernel ------------------------------------
// Per wave: 16 edges.  rf/cf = [deg, h] @ ftW (+b); cos; z1 = relu([cos,rf,cf]@aW+b);
// z2 = relu(z1@mW+b); s = sigmoid(z2 . cW + cB).
#define COT_WAVES 4
__global__ void __launch_bounds__(128)
cot_kernel(const float* __restrict__ h, const int* __restrict__ row,
           const int* __restrict__ col, const float* __restrict__ degS,
           const _Float16* __restrict__ ftPack, const float* __restrict__ ftW,
           const float* __restrict__ ftB, const _Float16* __restrict__ aRPack,
           const _Float16* __restrict__ aCPack, const float* __restrict__ aW,
           const float* __restrict__ aB, const _Float16* __restrict__ mPack,
           const float* __restrict__ mB, const float* __restrict__ cW,
           const float* __restrict__ cB, float* __restrict__ sOut) {
  __shared__ float rf[COT_WAVES][16][64];
  __shared__ float cf[COT_WAVES][16][64];
  __shared__ float z1[COT_WAVES][16][64];
  __shared__ float cosb[COT_WAVES][16];
  __shared__ float dgr[COT_WAVES][16];
  __shared__ float dgc[COT_WAVES][16];
  __shared__ int ridx[COT_WAVES][16];
  __shared__ int cidx[COT_WAVES][16];

  int wv = threadIdx.x >> 5;
  int lane = threadIdx.x & 31;
  int m = lane & 15, sel = lane >> 4;
  int tile = blockIdx.x * COT_WAVES + wv;
  int ebase = tile * 16;

  if (lane < 16) {
    int e = ebase + lane;
    int r = row[e], c = col[e];
    ridx[wv][lane] = r;
    cidx[wv][lane] = c;
    dgr[wv][lane] = degS[r];
    dgc[wv][lane] = degS[c];
  }
  __syncthreads();

  // GEMM1: rf / cf = h[idx] @ ftW[1:129] + deg*ftW[0] + ftB   (K=128, N=64)
  for (int src = 0; src < 2; ++src) {
    const int* idx = src ? cidx[wv] : ridx[wv];
    const float* dg = src ? dgc[wv] : dgr[wv];
    float* dst = src ? &cf[wv][0][0] : &rf[wv][0][0];
    const float* hrow = h + (size_t)idx[m] * HDIM;
    v16h a0 = load_a_global(hrow + 0, 1.0f);
    v16h a1 = load_a_global(hrow + 32, 1.0f);
    v16h a2 = load_a_global(hrow + 64, 1.0f);
    v16h a3 = load_a_global(hrow + 96, 1.0f);
    for (int nt = 0; nt < 4; ++nt) {
      v8f c = v8zero();
      c = wmma16(a0, load_b_pack(ftPack, 0, nt, 4), c);
      c = wmma16(a1, load_b_pack(ftPack, 1, nt, 4), c);
      c = wmma16(a2, load_b_pack(ftPack, 2, nt, 4), c);
      c = wmma16(a3, load_b_pack(ftPack, 3, nt, 4), c);
      int n = nt * 16 + (lane & 15);
      float w0 = ftW[n];  // row 0 of (129x64) = deg weight
      float bb = ftB[n];
#pragma unroll
      for (int r = 0; r < 8; ++r) {
        int mm = r + 8 * sel;
        dst[mm * 64 + n] = c[r] + dg[mm] * w0 + bb;
      }
    }
  }
  __syncthreads();

  // cosine
  if (lane < 16) {
    float dot = 0.0f, n1 = 0.0f, n2 = 0.0f;
    for (int j = 0; j < 64; ++j) {
      float a = rf[wv][lane][j], b = cf[wv][lane][j];
      dot += a * b;
      n1 += a * a;
      n2 += b * b;
    }
    n1 = fmaxf(sqrtf(n1), 1e-8f);
    n2 = fmaxf(sqrtf(n2), 1e-8f);
    cosb[wv][lane] = dot / (n1 * n2);
  }
  __syncthreads();

  // GEMM2: z1 = relu(rf@aW[1:65] + cf@aW[65:129] + cos*aW[0] + aB)  (N=64)
  {
    const float* rrow = rf[wv][m];
    const float* crow = cf[wv][m];
    v16h ar0 = load_a_lds(rrow, 0), ar1 = load_a_lds(rrow, 32);
    v16h ac0 = load_a_lds(crow, 0), ac1 = load_a_lds(crow, 32);
    for (int nt = 0; nt < 4; ++nt) {
      v8f c = v8zero();
      c = wmma16(ar0, load_b_pack(aRPack, 0, nt, 4), c);
      c = wmma16(ar1, load_b_pack(aRPack, 1, nt, 4), c);
      c = wmma16(ac0, load_b_pack(aCPack, 0, nt, 4), c);
      c = wmma16(ac1, load_b_pack(aCPack, 1, nt, 4), c);
      int n = nt * 16 + (lane & 15);
      float w0 = aW[n];  // row 0 of (129x64) = cos weight
      float bb = aB[n];
#pragma unroll
      for (int r = 0; r < 8; ++r) {
        int mm = r + 8 * sel;
        z1[wv][mm][n] = fmaxf(c[r] + cosb[wv][mm] * w0 + bb, 0.0f);
      }
    }
  }
  __syncthreads();

  // GEMM3: z2 = relu(z1 @ mW + mB)  (K=64, N=32); reuse rf as z2 storage.
  {
    const float* zrow = z1[wv][m];
    v16h a0 = load_a_lds(zrow, 0), a1 = load_a_lds(zrow, 32);
    for (int nt = 0; nt < 2; ++nt) {
      v8f c = v8zero();
      c = wmma16(a0, load_b_pack(mPack, 0, nt, 2), c);
      c = wmma16(a1, load_b_pack(mPack, 1, nt, 2), c);
      int n = nt * 16 + (lane & 15);
      float bb = mB[n];
#pragma unroll
      for (int r = 0; r < 8; ++r) {
        int mm = r + 8 * sel;
        rf[wv][mm][n] = fmaxf(c[r] + bb, 0.0f);
      }
    }
  }
  __syncthreads();

  if (lane < 16) {
    float acc = cB[0];
    for (int j = 0; j < 32; ++j) acc += rf[wv][lane][j] * cW[j];
    sOut[ebase + lane] = 1.0f / (1.0f + expf(-acc));
  }
}

// ---------------- exact per-graph top-k (stable, tie-break by index) -------

__device__ __forceinline__ unsigned f2ord(float f) {
  unsigned u = __float_as_uint(f);
  return (u & 0x80000000u) ? ~u : (u | 0x80000000u);
}

__global__ void __launch_bounds__(256)
topk_kernel(const float* __restrict__ s, float* __restrict__ active,
            float* __restrict__ ew, const int* __restrict__ largestP, int kper) {
  __shared__ unsigned hist[256];
  __shared__ unsigned sh_prefix;
  __shared__ int sh_rem;
  __shared__ int scanbuf[256];
  __shared__ int sh_run;
  int tid = threadIdx.x;
  long long base = (long long)blockIdx.x * EPG;
  int largest = *largestP;

  unsigned prefix = 0;
  int remaining = kper;
  for (int pass = 0; pass < 4; ++pass) {
    int shift = 24 - pass * 8;
    unsigned hmask = (pass == 0) ? 0u : (0xFFFFFFFFu << (shift + 8));
    hist[tid] = 0;
    __syncthreads();
    for (int e = tid; e < EPG; e += 256) {
      unsigned k = 0;
      if (active[base + e] > 0.0f) {
        float v = s[base + e];
        if (!largest) v = -v;
        k = f2ord(v);
      }
      if ((k & hmask) == prefix) atomicAdd(&hist[(k >> shift) & 255], 1u);
    }
    __syncthreads();
    if (tid == 0) {
      int cum = 0, b = 0;
      for (int bb = 255; bb >= 0; --bb) {
        int c = (int)hist[bb];
        if (cum + c >= remaining) { b = bb; break; }
        cum += c;
      }
      sh_prefix = prefix | ((unsigned)b << shift);
      sh_rem = remaining - cum;
    }
    __syncthreads();
    prefix = sh_prefix;
    remaining = sh_rem;
    __syncthreads();
  }
  unsigned T = prefix;
  int meq = remaining;  // how many key==T to keep (lowest indices first)
  if (tid == 0) sh_run = 0;
  __syncthreads();

  for (int chunk = 0; chunk < EPG; chunk += 256) {
    int e = chunk + tid;
    float sv = s[base + e];
    unsigned k = 0;
    if (active[base + e] > 0.0f) {
      float v = largest ? sv : -sv;
      k = f2ord(v);
    }
    int isEq = (k == T) ? 1 : 0;
    scanbuf[tid] = isEq;
    __syncthreads();
    for (int off = 1; off < 256; off <<= 1) {
      int t = (tid >= off) ? scanbuf[tid - off] : 0;
      __syncthreads();
      scanbuf[tid] += t;
      __syncthreads();
    }
    int excl = scanbuf[tid] - isEq;
    int runbase = sh_run;
    int selected = (k > T) || (isEq && (runbase + excl) < meq);
    float na = selected ? 1.0f : 0.0f;
    active[base + e] = na;
    ew[base + e] = sv * na;
    __syncthreads();
    if (tid == 0) sh_run += scanbuf[255];
    __syncthreads();
  }
}

__global__ void finalize_ew_kernel(const float* __restrict__ active,
                                   float* __restrict__ ew,
                                   const int* __restrict__ largestP, int E) {
  int e = blockIdx.x * blockDim.x + threadIdx.x;
  if (e >= E) return;
  if (*largestP == 0) ew[e] = (active[e] > 0.0f) ? (1.0f - ew[e]) : 0.0f;
}

__global__ void readout_kernel(const float* __restrict__ h, float* __restrict__ hg) {
  int i = blockIdx.x * blockDim.x + threadIdx.x;
  if (i >= NGRAPH * HDIM) return;
  int b = i >> 7, c = i & 127;
  const float* p = h + (size_t)b * 256 * HDIM + c;
  float s = 0.0f;
  for (int r = 0; r < 256; ++r) s += p[r * HDIM];
  hg[i] = s * (1.0f / 256.0f);
}

// ---------------------------------------------------------------------------

extern "C" void kernel_launch(void* const* d_in, const int* in_sizes, int n_in,
                              void* d_out, int out_size, void* d_ws, size_t ws_size,
                              hipStream_t stream) {
  (void)in_sizes; (void)n_in; (void)out_size; (void)ws_size;
  const float* x = (const float*)d_in[0];
  const int* row = (const int*)d_in[1];
  const int* col = (const int*)d_in[2];
  const int* largestP = (const int*)d_in[3];
  auto F = [&](int i) { return (const float*)d_in[i]; };

  struct LinP { const float *W, *b; };
  struct GcnP { const float *W1, *W2, *b1, *b2, *bt, *g; };
  struct CotP { LinP a, c, ft, m; };
  CotP cot[2]; GcnP gcn[2];
  for (int blk = 0; blk < 2; ++blk) {
    int b = 4 + blk * 14;
    cot[blk].a  = {F(b + 0), F(b + 1)};
    cot[blk].c  = {F(b + 2), F(b + 3)};
    cot[blk].ft = {F(b + 4), F(b + 5)};
    cot[blk].m  = {F(b + 6), F(b + 7)};
    gcn[blk] = {F(b + 8), F(b + 9), F(b + 10), F(b + 11), F(b + 12), F(b + 13)};
  }
  LinP causal = {F(32), F(33)};
  GcnP fin = {F(34), F(35), F(36), F(37), F(38), F(39)};

  // ---- workspace carving ----
  size_t off = 0;
  auto carve = [&](size_t bytes) -> void* {
    off = (off + 255) & ~(size_t)255;
    void* p = (char*)d_ws + off;
    off += bytes;
    return p;
  };
  float* hA   = (float*)carve((size_t)NNODES * HDIM * 4);
  float* t1   = (float*)carve((size_t)NNODES * HDIM * 4);
  float* t2   = (float*)carve((size_t)NNODES * HDIM * 4);
  float* acc  = (float*)carve((size_t)NNODES * HDIM * 4);
  float* act  = (float*)carve((size_t)NEDGE * 4);
  float* ewv  = (float*)carve((size_t)NEDGE * 4);
  float* degO = (float*)carve(NNODES * 4);
  float* degI = (float*)carve(NNODES * 4);
  float* invO = (float*)carve(NNODES * 4);
  float* invI = (float*)carve(NNODES * 4);
  float* degS = (float*)carve(NNODES * 4);
  float* mu   = (float*)carve(HDIM * 4);
  float* var  = (float*)carve(HDIM * 4);
  _Float16* pkCausal = (_Float16*)carve(8192 * 2);
  _Float16 *pkW1[3], *pkW2[3], *pkFt[2], *pkAR[2], *pkAC[2], *pkM[2];
  for (int i = 0; i < 3; ++i) {
    pkW1[i] = (_Float16*)carve(16384 * 2);
    pkW2[i] = (_Float16*)carve(16384 * 2);
  }
  for (int i = 0; i < 2; ++i) {
    pkFt[i] = (_Float16*)carve(8192 * 2);
    pkAR[i] = (_Float16*)carve(4096 * 2);
    pkAC[i] = (_Float16*)carve(4096 * 2);
    pkM[i]  = (_Float16*)carve(2048 * 2);
  }

  // ---- pack weights into WMMA B-fragment layout ----
  auto pack = [&](const float* W, int ldn, int rowOff, int K, int Nn,
                  _Float16* outp) {
    int total = (K / 32) * (Nn / 16) * 512;
    pack_b_kernel<<<(total + 255) / 256, 256, 0, stream>>>(W, ldn, rowOff, K, Nn,
                                                           outp);
  };
  pack(causal.W, 128, 0, 64, 128, pkCausal);
  for (int i = 0; i < 2; ++i) {
    pack(gcn[i].W1, 128, 0, 128, 128, pkW1[i]);
    pack(gcn[i].W2, 128, 0, 128, 128, pkW2[i]);
    pack(cot[i].ft.W, 64, 1, 128, 64, pkFt[i]);   // ft.W rows 1..128
    pack(cot[i].a.W, 64, 1, 64, 64, pkAR[i]);     // a.W rows 1..64 (rf part)
    pack(cot[i].a.W, 64, 65, 64, 64, pkAC[i]);    // a.W rows 65..128 (cf part)
    pack(cot[i].m.W, 32, 0, 64, 32, pkM[i]);
  }
  pack(fin.W1, 128, 0, 128, 128, pkW1[2]);
  pack(fin.W2, 128, 0, 128, 128, pkW2[2]);

  const long long NH = (long long)NNODES * HDIM;
  auto fillf = [&](float* p, float v, long long n) {
    fill_kernel<<<(unsigned)((n + 255) / 256), 256, 0, stream>>>(p, v, n);
  };

  // active = 1, ew = 1
  fillf(act, 1.0f, NEDGE);
  fillf(ewv, 1.0f, NEDGE);

  // h0 = x @ Wc + bc
  gemm_node_kernel<<<1024, 256, 0, stream>>>(x, nullptr, pkCausal, causal.b,
                                             nullptr, hA, 64, 0);

  auto gcn_block = [&](const GcnP& g, const _Float16* pw1, const _Float16* pw2) {
    fillf(degO, 0.0f, NNODES);
    fillf(degI, 0.0f, NNODES);
    deg_kernel<<<NEDGE / 256, 256, 0, stream>>>(row, col, act, degO, degI, NEDGE);
    invdeg_kernel<<<NNODES / 256, 256, 0, stream>>>(degO, degI, invO, invI, degS,
                                                    NNODES);
    // conv1 + relu
    fillf(acc, 0.0f, NH);
    edge_agg_kernel<<<(unsigned)((long long)NEDGE * 32 / 256), 256, 0, stream>>>(
        hA, row, col, ewv, invO, acc, NEDGE);
    gemm_node_kernel<<<1024, 256, 0, stream>>>(acc, invI, pw1, g.b1, nullptr, t1,
                                               128, 1);
    // conv2 + relu + residual
    fillf(acc, 0.0f, NH);
    edge_agg_kernel<<<(unsigned)((long long)NEDGE * 32 / 256), 256, 0, stream>>>(
        t1, row, col, ewv, invO, acc, NEDGE);
    gemm_node_kernel<<<1024, 256, 0, stream>>>(acc, invI, pw2, g.b2, hA, t2, 128,
                                               3);
    // batchnorm (training stats) -> hA
    bn_stats_kernel<<<128, 256, 0, stream>>>(t2, mu, var);
    bn_apply_kernel<<<(unsigned)(NH / 256), 256, 0, stream>>>(t2, mu, var, g.g,
                                                              g.bt, hA, NH);
  };

  float* hg = (float*)d_out;
  float* s0 = hg + NGRAPH * HDIM;
  float* s1 = s0 + NEDGE;
  const int kper[2] = {6553, 3931};  // int(8192*0.8), int(6553*0.6)

  for (int i = 0; i < 2; ++i) {
    gcn_block(gcn[i], pkW1[i], pkW2[i]);
    float* sOut = (i == 0) ? s0 : s1;
    cot_kernel<<<NEDGE / (16 * COT_WAVES), 128, 0, stream>>>(
        hA, row, col, degS, pkFt[i], cot[i].ft.W, cot[i].ft.b, pkAR[i], pkAC[i],
        cot[i].a.W, cot[i].a.b, pkM[i], cot[i].m.b, cot[i].c.W, cot[i].c.b, sOut);
    topk_kernel<<<NGRAPH, 256, 0, stream>>>(sOut, act, ewv, largestP, kper[i]);
  }

  finalize_ew_kernel<<<NEDGE / 256, 256, 0, stream>>>(act, ewv, largestP, NEDGE);
  gcn_block(fin, pkW1[2], pkW2[2]);
  readout_kernel<<<32, 256, 0, stream>>>(hA, hg);
}
